// MHA_87849261072413
// MI455X (gfx1250) — matrix operand order repaired and checked by reading, not compiled, not run on hardware
//
#include <hip/hip_runtime.h>
#include <stdint.h>

// ---------- types ----------
typedef __attribute__((ext_vector_type(16))) __bf16 v16bf;
typedef __attribute__((ext_vector_type(8)))  __bf16 v8bf;
typedef __attribute__((ext_vector_type(8)))  float  v8f;
typedef __attribute__((ext_vector_type(4)))  int    v4i;

union BF16Frag { v16bf v; v8bf h[2]; };

__device__ __forceinline__ uint16_t f2bf(float f) {
    union { float f; uint32_t u; } c; c.f = f;
    uint32_t u = c.u;
    u += 0x7FFFu + ((u >> 16) & 1u);          // round-to-nearest-even
    return (uint16_t)(u >> 16);
}

#define WMMA_BF16(A, B, C) \
    __builtin_amdgcn_wmma_f32_16x16x32_bf16(false, (A), false, (B), (short)0, (C), false, false)

// problem constants
#define BATCH 4
#define SEQ   2048
#define DMODEL 1024
#define NH    16
#define DH    64
#define MTOT  (BATCH * SEQ)        // 8192

// ---------- async copy global(16B) -> LDS, gfx1250 async path with fallback ----------
__device__ __forceinline__ void async_copy16(const uint16_t* g, uint16_t* l) {
#if __has_builtin(__builtin_amdgcn_global_load_async_to_lds_b128)
    __builtin_amdgcn_global_load_async_to_lds_b128(
        (__attribute__((address_space(1))) v4i*)(g),
        (__attribute__((address_space(3))) v4i*)(l), 0, 0);
#else
    *(v8bf*)l = *(const v8bf*)g;               // sync fallback: load + ds_store
#endif
}

__device__ __forceinline__ void wait_async0() {
#if __has_builtin(__builtin_amdgcn_s_wait_asynccnt)
    __builtin_amdgcn_s_wait_asynccnt(0);
#else
    asm volatile("s_wait_asynccnt 0" ::: "memory");
#endif
}

// ---------- fp32 -> bf16 conversion ----------
__global__ void cvt_bf16_kernel(const float* __restrict__ in, uint16_t* __restrict__ out, int n) {
    int i = blockIdx.x * blockDim.x + threadIdx.x;
    if (i < n) out[i] = f2bf(in[i]);
}

// ---------- QKV GEMM: C[m,n] = sum_k xb[m,k] * wb[n,k] + bias[n] ----------
__global__ __launch_bounds__(128) void gemm_qkv_kernel(
    const uint16_t* __restrict__ xb,   // [8192,1024] bf16
    const uint16_t* __restrict__ wb,   // [3072,1024] bf16
    const float*    __restrict__ bias, // [3072]
    uint16_t* __restrict__ Qb, uint16_t* __restrict__ Kb, uint16_t* __restrict__ Vt)
{
    const int lane   = threadIdx.x & 31;
    const int wave   = threadIdx.x >> 5;
    const int lane16 = lane & 15;
    const int half   = lane >> 4;

    const int NT = 3072 / 64;                       // 48 n-tiles
    const int tile = blockIdx.x * 4 + wave;
    const int tm = tile / NT, tn = tile % NT;
    const int m0 = tm * 64, n0 = tn * 64;

    v8f acc[4][4];
    #pragma unroll
    for (int i = 0; i < 4; ++i)
        #pragma unroll
        for (int j = 0; j < 4; ++j) acc[i][j] = (v8f){};

    for (int kc = 0; kc < DMODEL / 32; ++kc) {
        BF16Frag a[4], b[4];
        #pragma unroll
        for (int i = 0; i < 4; ++i) {
            const uint16_t* rA = xb + (size_t)(m0 + 16 * i + lane16) * DMODEL + kc * 32;
            a[i].h[0] = *(const v8bf*)(rA + 16 * half);
            a[i].h[1] = *(const v8bf*)(rA + 16 * half + 8);
        }
        #pragma unroll
        for (int j = 0; j < 4; ++j) {
            const uint16_t* rB = wb + (size_t)(n0 + 16 * j + lane16) * DMODEL + kc * 32 + 16 * half;
            b[j].h[0] = *(const v8bf*)(rB);
            b[j].h[1] = *(const v8bf*)(rB + 8);
        }
        #pragma unroll
        for (int i = 0; i < 4; ++i)
            #pragma unroll
            for (int j = 0; j < 4; ++j)
                acc[i][j] = WMMA_BF16(a[i].v, b[j].v, acc[i][j]);
    }

    #pragma unroll
    for (int i = 0; i < 4; ++i) {
        #pragma unroll
        for (int j = 0; j < 4; ++j) {
            #pragma unroll
            for (int r = 0; r < 8; ++r) {
                int m = m0 + 16 * i + 8 * half + r;      // token
                int n = n0 + 16 * j + lane16;            // output feature 0..3071
                float v = acc[i][j][r] + bias[n];
                int which = n >> 10;                     // 0=Q 1=K 2=V
                int col = n & 1023;
                int hh = col >> 6, dd = col & 63;
                int bb = m >> 11, ss = m & 2047;
                uint16_t bv = f2bf(v);
                if (which == 0)
                    Qb[((size_t)(bb * NH + hh) * SEQ + ss) * DH + dd] = bv;
                else if (which == 1)
                    Kb[((size_t)(bb * NH + hh) * SEQ + ss) * DH + dd] = bv;
                else
                    Vt[((size_t)(bb * NH + hh) * DH + dd) * SEQ + ss] = bv;
            }
        }
    }
}

// ---------- flash attention v2 ----------
// Block = 4 waves, one (b,h); each wave owns 32 query rows (block: 128 rows).
// K/V k-chunks (32 keys) staged in double-buffered LDS via async-to-LDS, shared by all waves.
#define KROW 72   // padded K row (halves): 144B, 16B-aligned, conflict-breaking
#define VROW 40   // padded V row (halves): 80B
__global__ __launch_bounds__(128) void attn_kernel(
    const uint16_t* __restrict__ Qb, const uint16_t* __restrict__ Kb,
    const uint16_t* __restrict__ Vt, uint16_t* __restrict__ o /* [8192,1024] bf16 */)
{
    __shared__ uint16_t Klds[2][32 * KROW];
    __shared__ uint16_t Vlds[2][64 * VROW];
    __shared__ uint16_t Plds[4][16 * VROW];

    const int tid    = threadIdx.x;
    const int lane   = tid & 31;
    const int wave   = tid >> 5;
    const int lane16 = lane & 15;
    const int half   = lane >> 4;

    const int bh   = blockIdx.x >> 4;                  // 0..63
    const int qblk = blockIdx.x & 15;                  // 16 q-blocks of 128 rows
    const int b    = bh >> 4, h = bh & 15;
    const int q0   = qblk * 128 + wave * 32;

    const uint16_t* Kbase = Kb + (size_t)bh * SEQ * DH;
    const uint16_t* Vbase = Vt + (size_t)bh * DH * SEQ;

    // Q A-fragments: 2 m-tiles x 2 d-chunks
    BF16Frag qf[2][2];
    #pragma unroll
    for (int mt = 0; mt < 2; ++mt) {
        const uint16_t* qrow = Qb + ((size_t)bh * SEQ + q0 + mt * 16 + lane16) * DH;
        #pragma unroll
        for (int c = 0; c < 2; ++c) {
            qf[mt][c].h[0] = *(const v8bf*)(qrow + c * 32 + 16 * half);
            qf[mt][c].h[1] = *(const v8bf*)(qrow + c * 32 + 16 * half + 8);
        }
    }

    float mrow[2][8], lrow[2][8];
    v8f accO[2][4];
    #pragma unroll
    for (int mt = 0; mt < 2; ++mt) {
        #pragma unroll
        for (int r = 0; r < 8; ++r) { mrow[mt][r] = -INFINITY; lrow[mt][r] = 0.f; }
        #pragma unroll
        for (int t = 0; t < 4; ++t) accO[mt][t] = (v8f){};
    }

    uint16_t* pl = &Plds[wave][0];

    // stage K (32x64) and V (64x32) chunk at key offset kb into buffer `buf`
    auto stage = [&](int buf, int kb) {
        #pragma unroll
        for (int it = 0; it < 2; ++it) {
            int idx = tid + it * 128;                  // 0..255 16B-chunks each
            int kr = idx >> 3, kc = (idx & 7) * 8;     // K: 32 rows x 64 halves
            async_copy16(Kbase + (size_t)(kb + kr) * DH + kc, &Klds[buf][kr * KROW + kc]);
            int vr = idx >> 2, vc = (idx & 3) * 8;     // V: 64 d-rows x 32 halves
            async_copy16(Vbase + (size_t)vr * SEQ + kb + vc, &Vlds[buf][vr * VROW + vc]);
        }
    };

    stage(0, 0);

    for (int kb = 0; kb < SEQ; kb += 32) {
        const int buf = (kb >> 5) & 1;
        wait_async0();                                 // my async fills done
        __syncthreads();                               // everyone's fills visible
        if (kb + 32 < SEQ) stage(buf ^ 1, kb + 32);    // prefetch next chunk

        // K B-fragments (2 key-col tiles x 2 d-chunks) and V B-fragments (4 d-tiles)
        BF16Frag kf[2][2], vf[4];
        #pragma unroll
        for (int t = 0; t < 2; ++t)
            #pragma unroll
            for (int c = 0; c < 2; ++c) {
                const uint16_t* p = &Klds[buf][(16 * t + lane16) * KROW + c * 32 + 16 * half];
                kf[t][c].h[0] = *(const v8bf*)(p);
                kf[t][c].h[1] = *(const v8bf*)(p + 8);
            }
        #pragma unroll
        for (int t = 0; t < 4; ++t) {
            const uint16_t* p = &Vlds[buf][(16 * t + lane16) * VROW + 16 * half];
            vf[t].h[0] = *(const v8bf*)(p);
            vf[t].h[1] = *(const v8bf*)(p + 8);
        }

        #pragma unroll
        for (int mt = 0; mt < 2; ++mt) {
            // ---- S = Q K^T ----
            v8f s[2];
            #pragma unroll
            for (int t = 0; t < 2; ++t) {
                s[t] = (v8f){};
                #pragma unroll
                for (int c = 0; c < 2; ++c)
                    s[t] = WMMA_BF16(qf[mt][c].v, kf[t][c].v, s[t]);
            }
            // ---- online softmax ----
            #pragma unroll
            for (int r = 0; r < 8; ++r) {
                s[0][r] *= 0.125f; s[1][r] *= 0.125f;  // 1/sqrt(64)
                float t = fmaxf(s[0][r], s[1][r]);
                t = fmaxf(t, __shfl_xor(t, 1));
                t = fmaxf(t, __shfl_xor(t, 2));
                t = fmaxf(t, __shfl_xor(t, 4));
                t = fmaxf(t, __shfl_xor(t, 8));
                float mn = fmaxf(mrow[mt][r], t);
                float alpha = __expf(mrow[mt][r] - mn);
                mrow[mt][r] = mn;
                float p0 = __expf(s[0][r] - mn);
                float p1 = __expf(s[1][r] - mn);
                s[0][r] = p0; s[1][r] = p1;
                float sm = p0 + p1;
                sm += __shfl_xor(sm, 1);
                sm += __shfl_xor(sm, 2);
                sm += __shfl_xor(sm, 4);
                sm += __shfl_xor(sm, 8);
                lrow[mt][r] = lrow[mt][r] * alpha + sm;
                #pragma unroll
                for (int t4 = 0; t4 < 4; ++t4) accO[mt][t4][r] *= alpha;
            }
            // ---- P: C-layout -> LDS -> A-fragment layout ----
            #pragma unroll
            for (int r = 0; r < 8; ++r) {
                int row = r + 8 * half;
                pl[row * VROW + lane16]      = f2bf(s[0][r]);
                pl[row * VROW + 16 + lane16] = f2bf(s[1][r]);
            }
            asm volatile("s_wait_dscnt 0" ::: "memory");   // wave-local LDS RAW
            BF16Frag pa;
            {
                const uint16_t* pr = pl + lane16 * VROW;
                pa.h[0] = *(const v8bf*)(pr + 16 * half);
                pa.h[1] = *(const v8bf*)(pr + 16 * half + 8);
            }
            asm volatile("" ::: "memory");
            // ---- O += P V ----
            #pragma unroll
            for (int t = 0; t < 4; ++t)
                accO[mt][t] = WMMA_BF16(pa.v, vf[t].v, accO[mt][t]);
        }
    }

    // ---- normalize, store O into [b, s, h*64+d] (= GEMM-ready [8192,1024]) ----
    #pragma unroll
    for (int mt = 0; mt < 2; ++mt) {
        float rl[8];
        #pragma unroll
        for (int r = 0; r < 8; ++r) rl[r] = 1.0f / lrow[mt][r];
        #pragma unroll
        for (int t = 0; t < 4; ++t)
            #pragma unroll
            for (int r = 0; r < 8; ++r) {
                int srow = q0 + mt * 16 + r + 8 * half;
                o[((size_t)(b * SEQ + srow)) * DMODEL + h * DH + 16 * t + lane16] =
                    f2bf(accO[mt][t][r] * rl[r]);
            }
    }
}

// ---------- output projection: out[m,n] = sum_k o[m,k]*wout[n,k] + bias[n] (fp32) ----------
__global__ __launch_bounds__(128) void gemm_out_kernel(
    const uint16_t* __restrict__ ob,    // [8192,1024] bf16
    const uint16_t* __restrict__ wb,    // [1024,1024] bf16
    const float*    __restrict__ bias,  // [1024]
    float* __restrict__ out)            // [8192,1024] fp32
{
    const int lane   = threadIdx.x & 31;
    const int wave   = threadIdx.x >> 5;
    const int lane16 = lane & 15;
    const int half   = lane >> 4;

    const int NT = DMODEL / 64;                        // 16
    const int tile = blockIdx.x * 4 + wave;
    const int tm = tile / NT, tn = tile % NT;
    const int m0 = tm * 64, n0 = tn * 64;

    v8f acc[4][4];
    #pragma unroll
    for (int i = 0; i < 4; ++i)
        #pragma unroll
        for (int j = 0; j < 4; ++j) acc[i][j] = (v8f){};

    for (int kc = 0; kc < DMODEL / 32; ++kc) {
        BF16Frag a[4], b[4];
        #pragma unroll
        for (int i = 0; i < 4; ++i) {
            const uint16_t* rA = ob + (size_t)(m0 + 16 * i + lane16) * DMODEL + kc * 32;
            a[i].h[0] = *(const v8bf*)(rA + 16 * half);
            a[i].h[1] = *(const v8bf*)(rA + 16 * half + 8);
        }
        #pragma unroll
        for (int j = 0; j < 4; ++j) {
            const uint16_t* rB = wb + (size_t)(n0 + 16 * j + lane16) * DMODEL + kc * 32 + 16 * half;
            b[j].h[0] = *(const v8bf*)(rB);
            b[j].h[1] = *(const v8bf*)(rB + 8);
        }
        #pragma unroll
        for (int i = 0; i < 4; ++i)
            #pragma unroll
            for (int j = 0; j < 4; ++j)
                acc[i][j] = WMMA_BF16(a[i].v, b[j].v, acc[i][j]);
    }

    #pragma unroll
    for (int i = 0; i < 4; ++i)
        #pragma unroll
        for (int j = 0; j < 4; ++j)
            #pragma unroll
            for (int r = 0; r < 8; ++r) {
                int m = m0 + 16 * i + 8 * half + r;
                int n = n0 + 16 * j + lane16;
                out[(size_t)m * DMODEL + n] = acc[i][j][r] + bias[n];
            }
}

// ---------- host launcher ----------
extern "C" void kernel_launch(void* const* d_in, const int* in_sizes, int n_in,
                              void* d_out, int out_size, void* d_ws, size_t ws_size,
                              hipStream_t stream) {
    const float* x      = (const float*)d_in[0];   // [4,2048,1024]
    const float* qkv_w  = (const float*)d_in[1];   // [3072,1024]
    const float* qkv_b  = (const float*)d_in[2];   // [3072]
    const float* out_w  = (const float*)d_in[3];   // [1024,1024]
    const float* out_b  = (const float*)d_in[4];   // [1024]
    float* out = (float*)d_out;

    // workspace layout (uint16_t units, all 16B aligned)
    uint16_t* ws = (uint16_t*)d_ws;
    const size_t nX    = (size_t)MTOT * DMODEL;        // 8,388,608
    const size_t nWqkv = (size_t)3 * DMODEL * DMODEL;  // 3,145,728
    const size_t nWout = (size_t)DMODEL * DMODEL;      // 1,048,576
    uint16_t* xb    = ws;
    uint16_t* wqkvb = xb + nX;
    uint16_t* woutb = wqkvb + nWqkv;
    uint16_t* Qb    = woutb + nWout;
    uint16_t* Kb    = Qb + nX;
    uint16_t* Vt    = Kb + nX;
    uint16_t* ob    = Vt + nX;

    // 1) conversions
    cvt_bf16_kernel<<<(int)((nX + 255) / 256), 256, 0, stream>>>(x, xb, (int)nX);
    cvt_bf16_kernel<<<(int)((nWqkv + 255) / 256), 256, 0, stream>>>(qkv_w, wqkvb, (int)nWqkv);
    cvt_bf16_kernel<<<(int)((nWout + 255) / 256), 256, 0, stream>>>(out_w, woutb, (int)nWout);

    // 2) QKV projection: 128 x 48 tiles of 64x64, 4 waves/block
    gemm_qkv_kernel<<<(128 * 48) / 4, 128, 0, stream>>>(xb, wqkvb, qkv_b, Qb, Kb, Vt);

    // 3) attention: 64 (b,h) x 16 q-blocks of 128 rows, 4 waves/block
    attn_kernel<<<64 * 16, 128, 0, stream>>>(Qb, Kb, Vt, ob);

    // 4) output projection: 128 x 16 tiles, 4 waves/block
    gemm_out_kernel<<<(128 * 16) / 4, 128, 0, stream>>>(ob, woutb, out_b, out);
}
